// LSTMPendulum_38130719654529
// MI455X (gfx1250) — compile-verified, matrix operand order checked by reading, not run on hardware
//
#include <hip/hip_runtime.h>

// ---------------------------------------------------------------------------
// LSTM (H=50, IN=2, T=1024, B=2048) + final 50->2 FC for MI455X (gfx1250).
//
// One workgroup (4 wave32s) per 16-row batch tile; wave w owns gate w
// (0=i,1=f,2=g,3=o) and computes its [16,50] slice of gates = gx + h @ W_hh^T
// with v_wmma_f32_16x16x32_f16 (f16 in, f32 accumulate). W_hh^T is resident in
// VGPRs for the whole kernel. Gate nonlinearities are applied per-wave (in
// parallel across the 4 waves) before the barrier; the serial cell update then
// needs only one tanh(c) per element. c stays f32 in LDS (no error compounding
// over the 1024 sequential steps); h is f16 in LDS feeding the WMMA A operand
// via two ds_load_b128 per K-chunk (72-half padded rows -> conflict-free).
// ---------------------------------------------------------------------------

typedef _Float16 v16h __attribute__((ext_vector_type(16)));
typedef _Float16 v8h  __attribute__((ext_vector_type(8)));
typedef float    v8f  __attribute__((ext_vector_type(8)));
typedef float    v4f  __attribute__((ext_vector_type(4)));

#define HID   50
#define NIN   2
#define NOUT  2
#define BATCH 2048
#define TSEQ  1024
#define HP    52   // padded row for f32 gate/c arrays
#define HROW  72   // padded f16 h row (36 banks -> conflict-free b128 reads)

__device__ __forceinline__ float fast_sigmoid(float x) {
    // 1 / (1 + 2^(-x*log2(e)))  ->  v_exp_f32 + v_rcp_f32
    float e = __builtin_amdgcn_exp2f(-1.4426950408889634f * x);
    return __builtin_amdgcn_rcpf(1.0f + e);
}
__device__ __forceinline__ float fast_tanh(float x) {
    // tanh(x) = 2*sigmoid(2x) - 1
    float e = __builtin_amdgcn_exp2f(-2.8853900817779268f * x);
    return 2.0f * __builtin_amdgcn_rcpf(1.0f + e) - 1.0f;
}

__global__ __launch_bounds__(128)
void lstm_pendulum_kernel(const float* __restrict__ x,
                          const float* __restrict__ W_ih,
                          const float* __restrict__ W_hh,
                          const float* __restrict__ b_ih,
                          const float* __restrict__ b_hh,
                          const float* __restrict__ W_fc,
                          const float* __restrict__ b_fc,
                          float* __restrict__ out)
{
    __shared__ __align__(16) _Float16 h_lds[16 * HROW];        // h_{t-1}, f16
    __shared__ __align__(16) float    c_lds[16 * HP];          // c, f32
    __shared__ __align__(16) float    g_lds[4 * 16 * HP];      // ACTIVATED gates
    __shared__ __align__(16) float    x_lds[2][16 * NIN];      // double-buffered x tile

    const int tid  = threadIdx.x;
    const int w    = tid >> 5;        // wave id == gate id (0=i,1=f,2=g,3=o)
    const int l    = tid & 31;        // lane in wave
    const int grp  = l >> 4;          // lane group (0: lanes 0-15, 1: lanes 16-31)
    const int ln   = l & 15;
    const int bbase = blockIdx.x * 16;

    // ---- init LDS state (h=0, c=0; padding stays zero forever) ----
    for (int i = tid; i < 16 * HROW; i += 128) h_lds[i] = (_Float16)0.0f;
    for (int i = tid; i < 16 * HP;  i += 128) c_lds[i] = 0.0f;

    // ---- W_hh^T -> WMMA B operand registers (resident all kernel) ----
    // 16-bit B layout assumed as the N<->M mirror of the documented A layout:
    //   lane ln holds column n = ln; lanes 0-15 hold K = kc*32 + {0..7,16..23},
    //   lanes 16-31 hold K = kc*32 + {8..15,24..31}; packed pairs per VGPR.
    v16h Bt[2][4];
    #pragma unroll
    for (int kc = 0; kc < 2; ++kc) {
        #pragma unroll
        for (int nt = 0; nt < 4; ++nt) {
            const int n   = nt * 16 + ln;     // column within this gate (N)
            const int row = w * HID + n;      // row of W_hh [200][50]
            v16h b;
            #pragma unroll
            for (int i = 0; i < 16; ++i) {
                const int ko = (i < 8) ? (grp * 8 + i) : (16 + grp * 8 + (i - 8));
                const int k  = kc * 32 + ko;
                const float v = (n < HID && k < HID) ? W_hh[row * HID + k] : 0.0f;
                b[i] = (_Float16)v;
            }
            Bt[kc][nt] = b;
        }
    }

    // ---- per-lane input-projection constants (K=2 -> pure VALU) ----
    float wi0[4], wi1[4], bs[4];
    #pragma unroll
    for (int nt = 0; nt < 4; ++nt) {
        const int n  = nt * 16 + ln;
        const int gn = w * HID + n;
        const bool valid = (n < HID);
        wi0[nt] = valid ? W_ih[gn * NIN + 0] : 0.0f;
        wi1[nt] = valid ? W_ih[gn * NIN + 1] : 0.0f;
        bs[nt]  = valid ? (b_ih[gn] + b_hh[gn]) : 0.0f;
    }

    // ---- t-invariant index mapping for the serial cell update ----
    // 16*50 = 800 elements over 128 lanes -> 7 strided slots per lane.
    int  ugofs[7], uhofs[7];
    bool uval[7];
    #pragma unroll
    for (int it = 0; it < 7; ++it) {
        const int idx = tid + it * 128;
        uval[it] = (idx < 16 * HID);
        const int m = idx / HID, n = idx - m * HID;
        ugofs[it] = m * HP + n;      // offset within one gate/c plane
        uhofs[it] = m * HROW + n;    // offset within h plane
    }

    // preload x[t=0]
    if (tid < 32) {
        const int row = tid >> 1, cmp = tid & 1;
        x_lds[0][row * NIN + cmp] = x[((size_t)(bbase + row) * TSEQ) * NIN + cmp];
    }
    __syncthreads();

    for (int t = 0; t < TSEQ; ++t) {
        // stage next x tile into the other buffer; prefetch further ahead
        if (w == 0) {
            const int row = l >> 1, cmp = l & 1;
            if (t + 1 < TSEQ)
                x_lds[(t + 1) & 1][row * NIN + cmp] =
                    x[((size_t)(bbase + row) * TSEQ + (t + 1)) * NIN + cmp];
            if (t + 8 < TSEQ)   // gfx1250 global_prefetch_b8
                __builtin_prefetch(&x[((size_t)(bbase + row) * TSEQ + (t + 8)) * NIN], 0, 0);
        }

        // my 8 rows of x (m = grp*8 .. grp*8+7); LDS broadcast reads
        const float* xp = &x_lds[t & 1][grp * 8 * NIN];
        v4f xv[4];
        xv[0] = *(const v4f*)(xp + 0);
        xv[1] = *(const v4f*)(xp + 4);
        xv[2] = *(const v4f*)(xp + 8);
        xv[3] = *(const v4f*)(xp + 12);
        float xa[8], xb[8];
        #pragma unroll
        for (int r = 0; r < 8; ++r) {
            xa[r] = xv[r >> 1][(r & 1) * 2 + 0];
            xb[r] = xv[r >> 1][(r & 1) * 2 + 1];
        }

        // accumulators start at gx + (b_ih + b_hh), laid out as WMMA C tiles:
        // lane l, VGPR r  <->  element (M = r + 8*grp, N = nt*16 + ln)
        v8f acc[4];
        #pragma unroll
        for (int nt = 0; nt < 4; ++nt) {
            v8f a;
            #pragma unroll
            for (int r = 0; r < 8; ++r)
                a[r] = fmaf(xa[r], wi0[nt], fmaf(xb[r], wi1[nt], bs[nt]));
            acc[nt] = a;
        }

        // h @ W_hh^T : 2 K-chunks x 4 N-tiles of v_wmma_f32_16x16x32_f16
        #pragma unroll
        for (int kc = 0; kc < 2; ++kc) {
            // A operand: lane holds row M = ln, contiguous K runs -> 2x ds_load_b128
            const _Float16* hp0 = &h_lds[ln * HROW + kc * 32 + grp * 8];
            v8h alo = *(const v8h*)(hp0);
            v8h ahi = *(const v8h*)(hp0 + 16);
            v16h amat = __builtin_shufflevector(alo, ahi,
                0, 1, 2, 3, 4, 5, 6, 7, 8, 9, 10, 11, 12, 13, 14, 15);
            #pragma unroll
            for (int nt = 0; nt < 4; ++nt)
                acc[nt] = __builtin_amdgcn_wmma_f32_16x16x32_f16(
                    false, amat, false, Bt[kc][nt], (short)0, acc[nt], false, false);
        }

        // apply THIS gate's nonlinearity in-register (parallel across 4 waves):
        // waves 0,1,3 -> sigmoid; wave 2 -> tanh. Uniform per wave.
        if (w == 2) {
            #pragma unroll
            for (int nt = 0; nt < 4; ++nt)
                #pragma unroll
                for (int r = 0; r < 8; ++r)
                    acc[nt][r] = fast_tanh(acc[nt][r]);
        } else {
            #pragma unroll
            for (int nt = 0; nt < 4; ++nt)
                #pragma unroll
                for (int r = 0; r < 8; ++r)
                    acc[nt][r] = fast_sigmoid(acc[nt][r]);
        }

        // activated gates -> LDS (mask n >= 50)
        #pragma unroll
        for (int nt = 0; nt < 4; ++nt) {
            const int n = nt * 16 + ln;
            if (n < HID) {
                #pragma unroll
                for (int r = 0; r < 8; ++r) {
                    const int m = grp * 8 + r;
                    g_lds[(w * 16 + m) * HP + n] = acc[nt][r];
                }
            }
        }
        __syncthreads();   // gates visible; h_lds/x reads of this step complete

        // serial cell update: c = f*c + i*g ; h = o*tanh(c)  (one tanh/element)
        #pragma unroll
        for (int it = 0; it < 7; ++it) {
            if (uval[it]) {
                const int go = ugofs[it];
                const float i_ = g_lds[0 * 16 * HP + go];
                const float f_ = g_lds[1 * 16 * HP + go];
                const float g_ = g_lds[2 * 16 * HP + go];
                const float o_ = g_lds[3 * 16 * HP + go];
                const float c  = f_ * c_lds[go] + i_ * g_;
                c_lds[go]       = c;
                h_lds[uhofs[it]] = (_Float16)(o_ * fast_tanh(c));
            }
        }
        __syncthreads();   // h_{t} ready for next step's A operand
    }

    // ---- final FC: out[row][o] = h_last[row] . W_fc[o] + b_fc[o] ----
    if (tid < 32) {
        const int row = tid >> 1, o = tid & 1;
        float acc = b_fc[o];
        #pragma unroll 5
        for (int n = 0; n < HID; ++n)
            acc += (float)h_lds[row * HROW + n] * W_fc[o * HID + n];
        out[(size_t)(bbase + row) * NOUT + o] = acc;
    }
}

extern "C" void kernel_launch(void* const* d_in, const int* in_sizes, int n_in,
                              void* d_out, int out_size, void* d_ws, size_t ws_size,
                              hipStream_t stream) {
    const float* x    = (const float*)d_in[0];
    const float* W_ih = (const float*)d_in[1];
    const float* W_hh = (const float*)d_in[2];
    const float* b_ih = (const float*)d_in[3];
    const float* b_hh = (const float*)d_in[4];
    const float* W_fc = (const float*)d_in[5];
    const float* b_fc = (const float*)d_in[6];
    float* out = (float*)d_out;

    dim3 grid(BATCH / 16);   // 128 independent batch tiles
    dim3 block(128);         // 4 wave32s: one per LSTM gate
    lstm_pendulum_kernel<<<grid, block, 0, stream>>>(x, W_ih, W_hh, b_ih, b_hh,
                                                     W_fc, b_fc, out);
}
